// AttnHead_84361747628640
// MI455X (gfx1250) — compile-verified, matrix operand order checked by reading, not run on hardware
//
#include <hip/hip_runtime.h>
#include <hip/hip_fp16.h>

#define DIN  256
#define DOUT 128
#define ALPHA 0.2f

typedef __attribute__((ext_vector_type(2)))  float    v2f;
typedef __attribute__((ext_vector_type(8)))  float    v8f;
typedef __attribute__((ext_vector_type(16))) _Float16 v16h;

// ---------------- init: out = bias, zero segment buffers ----------------
__global__ void k_init(float* __restrict__ out, const float* __restrict__ bias,
                       unsigned* __restrict__ m_enc, float* __restrict__ s, int n) {
  long long i = (long long)blockIdx.x * blockDim.x + threadIdx.x;
  long long total = (long long)n * DOUT;
  if (i < total) out[i] = bias[i & (DOUT - 1)];
  if (i < n) { m_enc[i] = 0u; s[i] = 0.0f; }
}

// ---------------- GEMM: seq = feat @ W via WMMA ----------------
// One wave computes a 16x128 tile: 8 accumulators of 16x16 f32.
__device__ __forceinline__ int kidx16(int i, int half) {
  // f16 16x32 A/B fragment K index for v16h element i (ISA 7.12.2 layout)
  int v = i >> 1, e = i & 1;
  return ((v & 4) << 2) + 2 * (v & 3) + e + 8 * half;
}

__global__ void k_gemm(const float* __restrict__ feat, const float* __restrict__ W,
                       float* __restrict__ seq, int n) {
  const int lane = threadIdx.x & 31;
  const int wave = threadIdx.x >> 5;
  const int half = lane >> 4;   // 0: lanes 0-15, 1: lanes 16-31
  const int l16  = lane & 15;
  const int row0 = (blockIdx.x * 8 + wave) * 16;
  if (row0 >= n) return;        // wave-uniform exit; EXEC stays all-ones inside

  v8f acc[8];
  #pragma unroll
  for (int t = 0; t < 8; ++t)
    #pragma unroll
    for (int j = 0; j < 8; ++j) acc[t][j] = 0.0f;

  int arow_i = row0 + l16; if (arow_i > n - 1) arow_i = n - 1;   // clamp for partial tile
  const float* arow = feat + (size_t)arow_i * DIN;

#if __has_builtin(__builtin_amdgcn_wmma_f32_16x16x4_f32)
  // fp32 WMMA, K=4 per step: A 16x4 (v2f/lane), B 4x16 (v2f/lane)
  for (int k = 0; k < DIN; k += 4) {
    v2f a;
    a.x = arow[k + 2 * half];
    a.y = arow[k + 2 * half + 1];
    const float* b0 = W + (size_t)(k + 2 * half) * DOUT + l16;
    #pragma unroll
    for (int ct = 0; ct < 8; ++ct) {
      v2f b;
      b.x = b0[ct * 16];
      b.y = b0[ct * 16 + DOUT];
      acc[ct] = __builtin_amdgcn_wmma_f32_16x16x4_f32(
          false, a, false, b, (short)0, acc[ct], false, false);
    }
  }
#else
  // fallback: f16 inputs, f32 accumulate, K=32 per step
  for (int k = 0; k < DIN; k += 32) {
    v16h a;
    #pragma unroll
    for (int i = 0; i < 16; ++i) a[i] = (_Float16)arow[k + kidx16(i, half)];
    #pragma unroll
    for (int ct = 0; ct < 8; ++ct) {
      v16h b;
      #pragma unroll
      for (int i = 0; i < 16; ++i)
        b[i] = (_Float16)W[(size_t)(k + kidx16(i, half)) * DOUT + ct * 16 + l16];
      acc[ct] = __builtin_amdgcn_wmma_f32_16x16x32_f16(
          false, a, false, b, (short)0, acc[ct], false, false);
    }
  }
#endif

  // C/D layout: VGPR r holds M=r (lanes 0-15) and M=r+8 (lanes 16-31), N = l16
  #pragma unroll
  for (int ct = 0; ct < 8; ++ct)
    #pragma unroll
    for (int r = 0; r < 8; ++r) {
      int row = row0 + r + 8 * half;
      if (row < n) seq[(size_t)row * DOUT + ct * 16 + l16] = acc[ct][r];
    }
}

// ---------------- f1/f2: per-node attention scalars ----------------
__global__ void k_f1f2(const float* __restrict__ seq,
                       const float* __restrict__ al_w, const float* __restrict__ al_b,
                       const float* __restrict__ ar_w, const float* __restrict__ ar_b,
                       float* __restrict__ f1, float* __restrict__ f2, int n) {
  const int lane = threadIdx.x & 31;
  const int wave = threadIdx.x >> 5;
  const int row  = blockIdx.x * 8 + wave;
  if (row >= n) return;
  float4 v  = ((const float4*)(seq + (size_t)row * DOUT))[lane];
  float4 wa = ((const float4*)al_w)[lane];
  float4 wb = ((const float4*)ar_w)[lane];
  float p1 = v.x * wa.x + v.y * wa.y + v.z * wa.z + v.w * wa.w;
  float p2 = v.x * wb.x + v.y * wb.y + v.z * wb.z + v.w * wb.w;
  #pragma unroll
  for (int m = 16; m >= 1; m >>= 1) {
    p1 += __shfl_xor(p1, m, 32);
    p2 += __shfl_xor(p2, m, 32);
  }
  if (lane == 0) { f1[row] = p1 + al_b[0]; f2[row] = p2 + ar_b[0]; }
}

// ---------------- ordered-uint encoding for float atomicMax ----------------
__device__ __forceinline__ unsigned f2ord(float f) {
  unsigned u = __float_as_uint(f);
  return (u & 0x80000000u) ? ~u : (u | 0x80000000u);
}
__device__ __forceinline__ float ord2f(unsigned u) {
  return (u & 0x80000000u) ? __uint_as_float(u & 0x7FFFFFFFu) : __uint_as_float(~u);
}

// ---------------- edge logits + segment max ----------------
__global__ void k_logits(const int* __restrict__ row, const int* __restrict__ col,
                         const float* __restrict__ f1, const float* __restrict__ f2,
                         float* __restrict__ logits, unsigned* __restrict__ m_enc, int e) {
  int i = blockIdx.x * blockDim.x + threadIdx.x;
  if (i >= e) return;
  int r = row[i], c = col[i];
  float l = f1[r] + f2[c];
  l = (l > 0.0f) ? l : ALPHA * l;          // leaky_relu
  logits[i] = l;
  atomicMax(&m_enc[r], f2ord(l));
}

// ---------------- exp(logit - max) + segment sum (in-place over logits) ----
__global__ void k_expsum(const int* __restrict__ row, float* __restrict__ ev,
                         const unsigned* __restrict__ m_enc, float* __restrict__ s, int e) {
  int i = blockIdx.x * blockDim.x + threadIdx.x;
  if (i >= e) return;
  int r = row[i];
  float x = __expf(ev[i] - ord2f(m_enc[r]));
  ev[i] = x;
  atomicAdd(&s[r], x);
}

// ---------------- SpMM: out[row] += (ev/s[row]) * seq[col] ----------------
// One wave per edge; lane handles 4 contiguous columns (float4 gather).
__global__ void k_spmm(const int* __restrict__ row, const int* __restrict__ col,
                       const float* __restrict__ ev, const float* __restrict__ s,
                       const float* __restrict__ seq, float* __restrict__ out, int e) {
  const int lane = threadIdx.x & 31;
  const int wave = threadIdx.x >> 5;
  long long i = (long long)blockIdx.x * 8 + wave;
  if (i >= e) return;
  int r = row[i], c = col[i];
  float coef = ev[i] / s[r];
  float4 v = ((const float4*)(seq + (size_t)c * DOUT))[lane];
  float* o = out + (size_t)r * DOUT + lane * 4;
  atomicAdd(o + 0, coef * v.x);
  atomicAdd(o + 1, coef * v.y);
  atomicAdd(o + 2, coef * v.z);
  atomicAdd(o + 3, coef * v.w);
}

// ---------------- launch ----------------
extern "C" void kernel_launch(void* const* d_in, const int* in_sizes, int n_in,
                              void* d_out, int out_size, void* d_ws, size_t ws_size,
                              hipStream_t stream) {
  const float* feat = (const float*)d_in[0];
  const int*   row  = (const int*)d_in[1];
  const int*   col  = (const int*)d_in[2];
  const float* W    = (const float*)d_in[3];
  const float* al_w = (const float*)d_in[4];
  const float* al_b = (const float*)d_in[5];
  const float* ar_w = (const float*)d_in[6];
  const float* ar_b = (const float*)d_in[7];
  const float* bias = (const float*)d_in[8];
  float* out = (float*)d_out;

  const int n = in_sizes[0] / DIN;   // 100000
  const int e = in_sizes[1];         // 3200000

  char* ws = (char*)d_ws;
  size_t off = 0;
  auto carve = [&](size_t bytes) -> void* {
    void* p = ws + off;
    off = (off + bytes + 255) & ~(size_t)255;
    return p;
  };
  float*    seq   = (float*)carve((size_t)n * DOUT * sizeof(float)); // 51.2 MB
  float*    f1    = (float*)carve((size_t)n * sizeof(float));
  float*    f2    = (float*)carve((size_t)n * sizeof(float));
  unsigned* m_enc = (unsigned*)carve((size_t)n * sizeof(unsigned));
  float*    s     = (float*)carve((size_t)n * sizeof(float));
  float*    ev    = (float*)carve((size_t)e * sizeof(float));        // 12.8 MB
  (void)ws_size; (void)n_in; (void)out_size;

  long long tot = (long long)n * DOUT;
  k_init  <<<(unsigned)((tot + 255) / 256), 256, 0, stream>>>(out, bias, m_enc, s, n);
  k_gemm  <<<(unsigned)((n + 127) / 128),   256, 0, stream>>>(feat, W, seq, n);
  k_f1f2  <<<(unsigned)((n + 7) / 8),       256, 0, stream>>>(seq, al_w, al_b, ar_w, ar_b, f1, f2, n);
  k_logits<<<(unsigned)((e + 255) / 256),   256, 0, stream>>>(row, col, f1, f2, ev, m_enc, e);
  k_expsum<<<(unsigned)((e + 255) / 256),   256, 0, stream>>>(row, ev, m_enc, s, e);
  k_spmm  <<<(unsigned)((e + 7) / 8),       256, 0, stream>>>(row, col, ev, s, seq, out, e);
}
